// ROIViTExtractor_12395275616452
// MI455X (gfx1250) — matrix operand order repaired and checked by reading, not compiled, not run on hardware
//
#include <hip/hip_runtime.h>
#include <hip/hip_bf16.h>

typedef __attribute__((ext_vector_type(16))) _Float16 v16h;
typedef __attribute__((ext_vector_type(8)))  _Float16 h8;
typedef __attribute__((ext_vector_type(8)))  float    v8f;

#define IMGSZ 224
#define PATCH 14
#define EDIM 1024
#define NHEAD 16
#define DHEAD 64
#define LAYERS 24
#define FFDIM 4096
#define NROI 32
#define NSIDE 16
#define NPATCH 256
#define TTOK 257
#define EPS_LN 1e-5f

// ---------------------------------------------------------------------------
// fp32 (KxN row-major) -> f16 transposed (NxK). Lane = n (coalesced reads),
// each thread writes one contiguous h8 run along k.
// ---------------------------------------------------------------------------
__global__ void cvtT_f16_kernel(const float* __restrict__ s,
                                _Float16* __restrict__ d, int K, int N) {
  int g = blockIdx.x * 256 + threadIdx.x;
  int n = g % N;
  int k0 = (g / N) * 8;
  if (k0 >= K) return;
  h8 o;
#pragma unroll
  for (int j = 0; j < 8; ++j)
    o[j] = (_Float16)s[(size_t)(k0 + j) * N + n];
  *(h8*)&d[(size_t)n * K + k0] = o;
}

// ---------------------------------------------------------------------------
// Concatenate three E-length f32 bias vectors -> 3E buffer. grid 12x256.
// ---------------------------------------------------------------------------
__global__ void concat3_kernel(const float* __restrict__ a,
                               const float* __restrict__ b,
                               const float* __restrict__ c,
                               float* __restrict__ d) {
  int i = blockIdx.x * 256 + threadIdx.x;
  float v = (i < EDIM) ? a[i] : ((i < 2 * EDIM) ? b[i - EDIM] : c[i - 2 * EDIM]);
  d[i] = v;
}

// ---------------------------------------------------------------------------
// Patch embedding: conv(stride=14) + ReLU, plus cls row. 257 blocks.
// ---------------------------------------------------------------------------
__global__ void patch_embed_kernel(const float* __restrict__ frame,
                                   const float* __restrict__ conv_w,
                                   const float* __restrict__ cls_emb,
                                   float* __restrict__ x) {
  const int b = blockIdx.x;
  const int tid = threadIdx.x;
  if (b == 0) {
    for (int it = 0; it < 4; ++it) {
      int e = tid + it * 256;
      x[e] = cls_emb[e];
    }
    return;
  }
  const int p = b - 1;
  const int y0 = (p / NSIDE) * PATCH, x0 = (p % NSIDE) * PATCH;
  __shared__ float sPix[3 * PATCH * PATCH];  // 588
  for (int idx = tid; idx < 3 * PATCH * PATCH; idx += 256) {
    int c = idx / (PATCH * PATCH);
    int rem = idx % (PATCH * PATCH);
    sPix[idx] = frame[c * IMGSZ * IMGSZ + (y0 + rem / PATCH) * IMGSZ +
                      (x0 + rem % PATCH)];
  }
  __syncthreads();
  for (int it = 0; it < 4; ++it) {
    int e = tid + it * 256;
    const float* w = conv_w + (size_t)e * (3 * PATCH * PATCH);
    float sum = 0.f;
#pragma unroll 4
    for (int s = 0; s < 3 * PATCH * PATCH; ++s) sum += w[s] * sPix[s];
    x[(size_t)(1 + p) * EDIM + e] = fmaxf(sum, 0.f);
  }
}

// ---------------------------------------------------------------------------
// ROI overlap matrix Mi (32x256) and normalized patch bias pb (256).
// ---------------------------------------------------------------------------
__global__ void roi_bias_kernel(const float* __restrict__ bboxes,
                                float* __restrict__ Mi,
                                float* __restrict__ pb) {
  const int j = threadIdx.x;
  const float px1 = (float)((j % NSIDE) * PATCH);
  const float py1 = (float)((j / NSIDE) * PATCH);
  const float px2 = px1 + PATCH, py2 = py1 + PATCH;
  float sum = 0.f;
  for (int r = 0; r < NROI; ++r) {
    float bx1 = bboxes[r * 4 + 0], by1 = bboxes[r * 4 + 1];
    float bx2 = bboxes[r * 4 + 2], by2 = bboxes[r * 4 + 3];
    float iw = fmaxf(fminf(bx2, px2) - fmaxf(bx1, px1), 0.f);
    float ih = fmaxf(fminf(by2, py2) - fmaxf(by1, py1), 0.f);
    float v = iw * ih / (float)(PATCH * PATCH);
    Mi[r * NPATCH + j] = v;
    sum += v;
  }
  __shared__ float red[256];
  red[j] = sum;
  __syncthreads();
  for (int s = 128; s > 0; s >>= 1) {
    if (j < s) red[j] = fmaxf(red[j], red[j + s]);
    __syncthreads();
  }
  pb[j] = sum / (red[0] + 1e-6f);
}

// ---------------------------------------------------------------------------
// LayerNorm: one block per row; writes f16 output (feeds WMMA A-operand).
// ---------------------------------------------------------------------------
__global__ void ln_kernel(const float* __restrict__ x,
                          const float* __restrict__ sc,
                          const float* __restrict__ bi,
                          _Float16* __restrict__ h) {
  const int row = blockIdx.x;
  const int tid = threadIdx.x;
  const float* xr = x + (size_t)row * EDIM;
  float vals[4];
  float s = 0.f, sq = 0.f;
#pragma unroll
  for (int it = 0; it < 4; ++it) {
    float v = xr[tid + it * 256];
    vals[it] = v;
    s += v;
    sq += v * v;
  }
  __shared__ float rs[256], rq[256];
  rs[tid] = s; rq[tid] = sq;
  __syncthreads();
  for (int t = 128; t > 0; t >>= 1) {
    if (tid < t) { rs[tid] += rs[tid + t]; rq[tid] += rq[tid + t]; }
    __syncthreads();
  }
  const float mean = rs[0] * (1.f / EDIM);
  const float var = rq[0] * (1.f / EDIM) - mean * mean;
  const float inv = rsqrtf(var + EPS_LN);
  _Float16* hr = h + (size_t)row * EDIM;
#pragma unroll
  for (int it = 0; it < 4; ++it) {
    int e = tid + it * 256;
    hr[e] = (_Float16)((vals[it] - mean) * inv * sc[e] + bi[e]);
  }
}

// ---------------------------------------------------------------------------
// WMMA GEMM: C(MxN) = A16(MxK) @ Bt16(NxK, pre-transposed) + bias [+ epi]
// EPI: 0 = bias, 1 = bias + quick_gelu, 2 = bias + residual add (res f32)
// Block: 256 thr / 8 waves; tile 32(M) x 256(N); each wave owns 32x32 ->
// 2 A-frags x 2 B-frags (all ds_load_b128) -> 4 WMMA per K-step.
// Double-buffered LDS ping-pong: one barrier per K-step; next tile's global
// loads are issued before the barrier and retire under the WMMAs.
// LDS rows padded to 40 halves (80B = 20-bank stride, conflict-free frags).
// ---------------------------------------------------------------------------
template <int EPI, typename OutT>
__global__ void gemm_wmma_kernel(const _Float16* __restrict__ A,
                                 const _Float16* __restrict__ Bt,
                                 const float* __restrict__ bias,
                                 const float* __restrict__ res,
                                 OutT* __restrict__ C,
                                 int M, int N, int K) {
  __shared__ _Float16 sA[2][32][40];
  __shared__ _Float16 sBt[2][256][40];

  const int tid = threadIdx.x;
  const int wave = tid >> 5;
  const int lane = tid & 31;
  const int m = lane & 15;
  const int hi = lane >> 4;
  const int r0 = blockIdx.y * 32;
  const int cb = blockIdx.x * 256;

  // Per-thread staging offsets (32-bit, computed once; k-loop is add-only).
  const int aRow = tid >> 2;                 // threads 0..127 stage A
  const int aC8 = (tid & 3) * 8;
  const bool aOk = (tid < 128) && (r0 + aRow < M);
  const int aOff = (r0 + aRow) * K + aC8;

  int bOff[4], bCol[4], bCh[4];
#pragma unroll
  for (int i = 0; i < 4; ++i) {
    int f = tid + i * 256;
    bCol[i] = f >> 2;
    bCh[i] = (f & 3) * 8;
    bOff[i] = (cb + bCol[i]) * K + bCh[i];
  }

  v8f acc[2][2];
#pragma unroll
  for (int a = 0; a < 2; ++a)
#pragma unroll
    for (int b = 0; b < 2; ++b)
      acc[a][b] = (v8f){0.f, 0.f, 0.f, 0.f, 0.f, 0.f, 0.f, 0.f};

  auto compute = [&](int p) {
    const h8* pa0 = (const h8*)&sA[p][m][0];
    const h8* pa1 = (const h8*)&sA[p][16 + m][0];
    const h8* pb0 = (const h8*)&sBt[p][wave * 32 + m][0];
    const h8* pb1 = (const h8*)&sBt[p][wave * 32 + 16 + m][0];
    v16h a0 = __builtin_shufflevector(pa0[hi], pa0[2 + hi],
                                      0, 1, 2, 3, 4, 5, 6, 7,
                                      8, 9, 10, 11, 12, 13, 14, 15);
    v16h a1 = __builtin_shufflevector(pa1[hi], pa1[2 + hi],
                                      0, 1, 2, 3, 4, 5, 6, 7,
                                      8, 9, 10, 11, 12, 13, 14, 15);
    v16h b0 = __builtin_shufflevector(pb0[hi], pb0[2 + hi],
                                      0, 1, 2, 3, 4, 5, 6, 7,
                                      8, 9, 10, 11, 12, 13, 14, 15);
    v16h b1 = __builtin_shufflevector(pb1[hi], pb1[2 + hi],
                                      0, 1, 2, 3, 4, 5, 6, 7,
                                      8, 9, 10, 11, 12, 13, 14, 15);
    acc[0][0] = __builtin_amdgcn_wmma_f32_16x16x32_f16(false, a0, false, b0,
                                                       (short)0, acc[0][0],
                                                       false, false);
    acc[0][1] = __builtin_amdgcn_wmma_f32_16x16x32_f16(false, a0, false, b1,
                                                       (short)0, acc[0][1],
                                                       false, false);
    acc[1][0] = __builtin_amdgcn_wmma_f32_16x16x32_f16(false, a1, false, b0,
                                                       (short)0, acc[1][0],
                                                       false, false);
    acc[1][1] = __builtin_amdgcn_wmma_f32_16x16x32_f16(false, a1, false, b1,
                                                       (short)0, acc[1][1],
                                                       false, false);
  };

  // Prologue: stage tile 0 into buffer 0.
  {
    if (tid < 128) {
      h8 v = {0, 0, 0, 0, 0, 0, 0, 0};
      if (aOk) v = *(const h8*)&A[aOff];
      *(h8*)&sA[0][aRow][aC8] = v;
    }
#pragma unroll
    for (int i = 0; i < 4; ++i)
      *(h8*)&sBt[0][bCol[i]][bCh[i]] = *(const h8*)&Bt[bOff[i]];
  }

  int p = 0;
  int k0 = 0;
  for (; k0 < K - 32; k0 += 32) {
    // Issue next tile's global loads (retire under the WMMAs below).
    h8 na = {0, 0, 0, 0, 0, 0, 0, 0};
    if (aOk) na = *(const h8*)&A[aOff + k0 + 32];
    h8 nb0 = *(const h8*)&Bt[bOff[0] + k0 + 32];
    h8 nb1 = *(const h8*)&Bt[bOff[1] + k0 + 32];
    h8 nb2 = *(const h8*)&Bt[bOff[2] + k0 + 32];
    h8 nb3 = *(const h8*)&Bt[bOff[3] + k0 + 32];
    if (k0 + 64 < K) __builtin_prefetch(&Bt[bOff[0] + k0 + 64], 0, 1);

    __syncthreads();          // buffer p fully staged
    compute(p);
    // Store next tile into the other buffer (safe: p^1 was last read one
    // iteration ago, before the barrier above).
    if (tid < 128) *(h8*)&sA[p ^ 1][aRow][aC8] = na;
    *(h8*)&sBt[p ^ 1][bCol[0]][bCh[0]] = nb0;
    *(h8*)&sBt[p ^ 1][bCol[1]][bCh[1]] = nb1;
    *(h8*)&sBt[p ^ 1][bCol[2]][bCh[2]] = nb2;
    *(h8*)&sBt[p ^ 1][bCol[3]][bCh[3]] = nb3;
    p ^= 1;
  }
  __syncthreads();
  compute(p);

  // Epilogue.
#pragma unroll
  for (int ms = 0; ms < 2; ++ms) {
#pragma unroll
    for (int ns = 0; ns < 2; ++ns) {
      const int col = cb + wave * 32 + ns * 16 + m;
      const float bcol = bias[col];
#pragma unroll
      for (int i = 0; i < 8; ++i) {
        int row = r0 + ms * 16 + i + hi * 8;
        if (row < M) {
          float v = acc[ms][ns][i] + bcol;
          if (EPI == 1) {
            v = v / (1.f + __expf(-1.702f * v));  // quick_gelu
          } else if (EPI == 2) {
            v += res[(size_t)row * N + col];
          }
          C[(size_t)row * N + col] = (OutT)v;
        }
      }
    }
  }
}

// ---------------------------------------------------------------------------
// Attention over fused QKV buffer (rows of 3*EDIM: [q | k | v]).
// One block per (query row, head). Writes f16 (feeds Wo GEMM).
// ---------------------------------------------------------------------------
__global__ void attn_kernel(const float* __restrict__ qkv,
                            const float* __restrict__ pb,
                            _Float16* __restrict__ o) {
  const int row = blockIdx.x;
  const int head = blockIdx.y;
  const int tid = threadIdx.x;
  const float scale = 0.125f;  // 1/sqrt(64)
  const int LD = 3 * EDIM;

  __shared__ float sQ[DHEAD];
  __shared__ float sS[TTOK];
  __shared__ float red[256];

  if (tid < DHEAD) sQ[tid] = qkv[(size_t)row * LD + head * DHEAD + tid];
  __syncthreads();

  for (int j = tid; j < TTOK; j += 256) {
    const float* kr = qkv + (size_t)j * LD + EDIM + head * DHEAD;
    float dot = 0.f;
#pragma unroll 8
    for (int d = 0; d < DHEAD; ++d) dot += sQ[d] * kr[d];
    float sc = dot * scale;
    if (row == 0 && j > 0) sc += pb[j - 1];
    sS[j] = sc;
  }
  __syncthreads();

  float lmax = -3.4e38f;
  for (int j = tid; j < TTOK; j += 256) lmax = fmaxf(lmax, sS[j]);
  red[tid] = lmax;
  __syncthreads();
  for (int s = 128; s > 0; s >>= 1) {
    if (tid < s) red[tid] = fmaxf(red[tid], red[tid + s]);
    __syncthreads();
  }
  const float mx = red[0];
  __syncthreads();

  float lsum = 0.f;
  for (int j = tid; j < TTOK; j += 256) {
    float e = __expf(sS[j] - mx);
    sS[j] = e;
    lsum += e;
  }
  red[tid] = lsum;
  __syncthreads();
  for (int s = 128; s > 0; s >>= 1) {
    if (tid < s) red[tid] += red[tid + s];
    __syncthreads();
  }
  const float inv = 1.f / red[0];
  __syncthreads();

  if (tid < DHEAD) {
    float acc = 0.f;
    for (int j = 0; j < TTOK; ++j)
      acc += sS[j] * qkv[(size_t)j * LD + 2 * EDIM + head * DHEAD + tid];
    o[(size_t)row * EDIM + head * DHEAD + tid] = (_Float16)(acc * inv);
  }
}

// ---------------------------------------------------------------------------
// Output: row 0 = cls token; rows 1..32 = Mi @ patch_out.
// ---------------------------------------------------------------------------
__global__ void roi_out_kernel(const float* __restrict__ x,
                               const float* __restrict__ Mi,
                               float* __restrict__ out) {
  const int b = blockIdx.x;
  const int tid = threadIdx.x;
  if (b == 0) {
    for (int it = 0; it < 4; ++it) {
      int e = tid + it * 256;
      out[e] = x[e];
    }
    return;
  }
  const int r = b - 1;
  for (int it = 0; it < 4; ++it) {
    int e = tid + it * 256;
    float acc = 0.f;
    for (int j = 0; j < NPATCH; ++j)
      acc += Mi[r * NPATCH + j] * x[(size_t)(1 + j) * EDIM + e];
    out[(size_t)b * EDIM + e] = acc;
  }
}

// ---------------------------------------------------------------------------
// Launch
// ---------------------------------------------------------------------------
extern "C" void kernel_launch(void* const* d_in, const int* in_sizes, int n_in,
                              void* d_out, int out_size, void* d_ws, size_t ws_size,
                              hipStream_t stream) {
  const float* frame  = (const float*)d_in[0];
  const float* bboxes = (const float*)d_in[1];
  const float* conv_w = (const float*)d_in[2];
  const float* cls    = (const float*)d_in[3];
  const float* ln1_s  = (const float*)d_in[4];
  const float* ln1_b  = (const float*)d_in[5];
  const float* wq     = (const float*)d_in[6];
  const float* bq     = (const float*)d_in[7];
  const float* wk     = (const float*)d_in[8];
  const float* bk     = (const float*)d_in[9];
  const float* wv     = (const float*)d_in[10];
  const float* bv     = (const float*)d_in[11];
  const float* wo     = (const float*)d_in[12];
  const float* bo     = (const float*)d_in[13];
  const float* ln2_s  = (const float*)d_in[14];
  const float* ln2_b  = (const float*)d_in[15];
  const float* w1     = (const float*)d_in[16];
  const float* b1     = (const float*)d_in[17];
  const float* w2     = (const float*)d_in[18];
  const float* b2     = (const float*)d_in[19];
  float* out = (float*)d_out;

  // Workspace carve (256B aligned).
  char* wp = (char*)d_ws;
  auto alloc = [&](size_t bytes) {
    char* r = wp;
    wp += (bytes + 255) & ~(size_t)255;
    return r;
  };
  const size_t XSZ = (size_t)TTOK * EDIM;
  float*     x     = (float*)alloc(XSZ * 4);
  float*     qkv   = (float*)alloc(XSZ * 3 * 4);
  float*     Mi    = (float*)alloc((size_t)NROI * NPATCH * 4);
  float*     pb    = (float*)alloc(NPATCH * 4);
  float*     bqkv  = (float*)alloc(3 * EDIM * 4);
  _Float16*  h16   = (_Float16*)alloc(XSZ * 2);
  _Float16*  att16 = (_Float16*)alloc(XSZ * 2);
  _Float16*  m16   = (_Float16*)alloc((size_t)TTOK * FFDIM * 2);
  _Float16*  wqkv16= (_Float16*)alloc((size_t)3 * EDIM * EDIM * 2);
  _Float16*  wo16  = (_Float16*)alloc((size_t)EDIM * EDIM * 2);
  _Float16*  w116  = (_Float16*)alloc((size_t)EDIM * FFDIM * 2);
  _Float16*  w216  = (_Float16*)alloc((size_t)FFDIM * EDIM * 2);

  const int MT = (TTOK + 31) / 32;          // 9 row tiles of 32
  const int CB_EE = (EDIM * EDIM) / (256 * 8);
  const int CB_EF = (EDIM * FFDIM) / (256 * 8);
  const size_t WEE = (size_t)EDIM * EDIM;

  patch_embed_kernel<<<TTOK, 256, 0, stream>>>(frame, conv_w, cls, x);
  roi_bias_kernel<<<1, 256, 0, stream>>>(bboxes, Mi, pb);

  for (int l = 0; l < LAYERS; ++l) {
    const size_t oE  = (size_t)l * EDIM;
    const size_t oEE = (size_t)l * EDIM * EDIM;
    const size_t oEF = (size_t)l * EDIM * FFDIM;
    const size_t oF  = (size_t)l * FFDIM;

    // Per-layer weight conversion fp32 -> f16 with transpose (N x K).
    // Q, K, V land in one concatenated (3E x E) buffer for a fused GEMM.
    cvtT_f16_kernel<<<CB_EE, 256, 0, stream>>>(wq + oEE, wqkv16, EDIM, EDIM);
    cvtT_f16_kernel<<<CB_EE, 256, 0, stream>>>(wk + oEE, wqkv16 + WEE, EDIM, EDIM);
    cvtT_f16_kernel<<<CB_EE, 256, 0, stream>>>(wv + oEE, wqkv16 + 2 * WEE, EDIM, EDIM);
    cvtT_f16_kernel<<<CB_EE, 256, 0, stream>>>(wo + oEE, wo16, EDIM, EDIM);
    cvtT_f16_kernel<<<CB_EF, 256, 0, stream>>>(w1 + oEF, w116, EDIM, FFDIM);
    cvtT_f16_kernel<<<CB_EF, 256, 0, stream>>>(w2 + oEF, w216, FFDIM, EDIM);
    concat3_kernel<<<12, 256, 0, stream>>>(bq + oE, bk + oE, bv + oE, bqkv);

    ln_kernel<<<TTOK, 256, 0, stream>>>(x, ln1_s + oE, ln1_b + oE, h16);

    // Fused QKV projection: M=257, N=3072, K=1024 (108 blocks).
    gemm_wmma_kernel<0, float><<<dim3(3 * EDIM / 256, MT), 256, 0, stream>>>(
        h16, wqkv16, bqkv, nullptr, qkv, TTOK, 3 * EDIM, EDIM);

    attn_kernel<<<dim3(TTOK, NHEAD), 256, 0, stream>>>(qkv, pb, att16);

    gemm_wmma_kernel<2, float><<<dim3(EDIM / 256, MT), 256, 0, stream>>>(
        att16, wo16, bo + oE, x, x, TTOK, EDIM, EDIM);

    ln_kernel<<<TTOK, 256, 0, stream>>>(x, ln2_s + oE, ln2_b + oE, h16);

    gemm_wmma_kernel<1, _Float16><<<dim3(FFDIM / 256, MT), 256, 0, stream>>>(
        h16, w116, b1 + oF, nullptr, m16, TTOK, FFDIM, EDIM);

    gemm_wmma_kernel<2, float><<<dim3(EDIM / 256, MT), 256, 0, stream>>>(
        m16, w216, b2 + oE, x, x, TTOK, EDIM, FFDIM);
  }

  roi_out_kernel<<<1 + NROI, 256, 0, stream>>>(x, Mi, out);
}
